// xLSTMModel_76390288327652
// MI455X (gfx1250) — compile-verified, matrix operand order checked by reading, not run on hardware
//
#include <hip/hip_runtime.h>
#include <hip/hip_bf16.h>

// ---------------- model dims ----------------
#define VOC 1024
#define EMB 256
#define HID 512
#define BSZ 256
#define SEQ 256
#define NGATE 2560            // 5*H : i,f,g,o gates + e-gate (We)

// ---------------- GEMM tiling ----------------
#define BM 128
#define BN 128
#define BK 32
#define LDST 40               // padded LDS row stride in elements (80B -> conflict-free)

typedef __attribute__((ext_vector_type(16))) __bf16 v16bf;
typedef __attribute__((ext_vector_type(8)))  __bf16 v8bf;
typedef __attribute__((ext_vector_type(8)))  float  v8f;
typedef __attribute__((address_space(3)))    __bf16 lds_bf16;

// ------------------------------------------------------------------
// Fused bf16 GEMM:  C[M,N] = A[M,K] * Bw[N,K]^T + bias[N]
// A is split at column K0 between two source buffers (A0 | A1) so the
// LSTM step can consume [x_t | h_{t-1}] without a concat copy.
// grid = (N/BN, M/BM), block = 256 threads (8 wave32 waves, 2x4).
// Each wave computes a 64x32 tile = 4x2 v_wmma_f32_16x16x32_bf16.
// Tiles stream into double-buffered LDS via the CDNA5 async engine
// (global_load_async_to_lds_b128, ASYNCcnt) — no VGPR staging.
// ------------------------------------------------------------------
__global__ __launch_bounds__(256) void gemm_bf16_kernel(
    const __bf16* __restrict__ A0, int lda0, int K0,
    const __bf16* __restrict__ A1, int lda1,
    const __bf16* __restrict__ Bw, int ldb,
    const float*  __restrict__ bias,
    float* __restrict__ C, int ldc, int K)
{
    __shared__ __align__(16) __bf16 As[2][BM * LDST];
    __shared__ __align__(16) __bf16 Bs[2][BN * LDST];

    const int tid  = threadIdx.x;
    const int lane = tid & 31;
    const int wave = tid >> 5;
    const int wm   = wave >> 2;            // 0..1  (M direction)
    const int wn   = wave & 3;             // 0..3  (N direction)
    const int mbase = blockIdx.y * BM;
    const int nbase = blockIdx.x * BN;

    // cooperative tile staging: 2 threads per 32-elem row, 16 elems each
    const int lrow  = tid >> 1;            // 0..127
    const int lhalf = (tid & 1) * 16;      // 0 or 16 elements

    v8f acc[4][2];
#pragma unroll
    for (int i = 0; i < 4; ++i)
#pragma unroll
        for (int j = 0; j < 2; ++j)
            acc[i][j] = (v8f)(0.0f);

    const int kb   = (lane < 16) ? 0 : 8;  // per-lane K sub-offset (WMMA A/B layout)
    const int rsub = lane & 15;
    const int KT   = K / BK;

    // issue one 128x32 A tile + 128x32 B tile into LDS buffer `buf`
    auto issue_tile = [&](int kt, int buf) {
        const int kcol = kt * BK;
        const __bf16* Ap; int Ald, acol;
        if (kcol < K0) { Ap = A0; Ald = lda0; acol = kcol; }
        else           { Ap = A1; Ald = lda1; acol = kcol - K0; }
        const __bf16* ag = Ap + (size_t)(mbase + lrow) * Ald + acol + lhalf;
        const __bf16* bg = Bw + (size_t)(nbase + lrow) * ldb + kcol + lhalf;
        const unsigned la =
            (unsigned)(unsigned long)(lds_bf16*)&As[buf][lrow * LDST + lhalf];
        const unsigned lb =
            (unsigned)(unsigned long)(lds_bf16*)&Bs[buf][lrow * LDST + lhalf];
        // two b128 transfers per operand: IOFFSET applies to LDS + global alike
        asm volatile("global_load_async_to_lds_b128 %0, %1, off\n\t"
                     "global_load_async_to_lds_b128 %0, %1, off offset:16"
                     :: "v"(la), "v"(ag) : "memory");
        asm volatile("global_load_async_to_lds_b128 %0, %1, off\n\t"
                     "global_load_async_to_lds_b128 %0, %1, off offset:16"
                     :: "v"(lb), "v"(bg) : "memory");
        // stream tile kt+1 toward L2 while kt lands in LDS
        __builtin_prefetch(ag + BK, 0, 1);
        __builtin_prefetch(bg + BK, 0, 1);
    };

    issue_tile(0, 0);
    for (int kt = 0; kt < KT; ++kt) {
        const int cur = kt & 1;
        if (kt + 1 < KT) {
            issue_tile(kt + 1, cur ^ 1);
            // 8 outstanding; wait until current tile's 4 are complete
            asm volatile("s_wait_asynccnt 0x4" ::: "memory");
        } else {
            asm volatile("s_wait_asynccnt 0x0" ::: "memory");
        }
        __syncthreads();                 // whole tile visible to all waves

        v16bf fa[4], fb[2];
#pragma unroll
        for (int i = 0; i < 4; ++i) {
            const __bf16* p = &As[cur][(wm * 64 + i * 16 + rsub) * LDST + kb];
            v8bf lo = *(const v8bf*)p;
            v8bf hi = *(const v8bf*)(p + 16);
            fa[i] = __builtin_shufflevector(lo, hi, 0,1,2,3,4,5,6,7,8,9,10,11,12,13,14,15);
        }
#pragma unroll
        for (int j = 0; j < 2; ++j) {
            const __bf16* p = &Bs[cur][(wn * 32 + j * 16 + rsub) * LDST + kb];
            v8bf lo = *(const v8bf*)p;
            v8bf hi = *(const v8bf*)(p + 16);
            fb[j] = __builtin_shufflevector(lo, hi, 0,1,2,3,4,5,6,7,8,9,10,11,12,13,14,15);
        }
#pragma unroll
        for (int i = 0; i < 4; ++i)
#pragma unroll
            for (int j = 0; j < 2; ++j)
                acc[i][j] = __builtin_amdgcn_wmma_f32_16x16x32_bf16(
                    false, fa[i], false, fb[j], (short)0, acc[i][j], false, false);

        __syncthreads();                 // done reading buf before it is refilled
    }

    // epilogue: C/D layout — lanes 0-15: M=r, lanes 16-31: M=r+8; N = lane&15
    const int radd = (lane < 16) ? 0 : 8;
#pragma unroll
    for (int i = 0; i < 4; ++i)
#pragma unroll
        for (int j = 0; j < 2; ++j) {
            const int col  = nbase + wn * 32 + j * 16 + rsub;
            const float bc = bias[col];
#pragma unroll
            for (int r = 0; r < 8; ++r) {
                const int row = mbase + wm * 64 + i * 16 + r + radd;
                C[(size_t)row * ldc + col] = acc[i][j][r] + bc;
            }
        }
}

// ------------------------------------------------------------------
// xLSTM pointwise: gates -> (c, h);  h emitted as bf16 for next GEMMs
// G layout per row b: [i(512) | f(512) | g(512) | o(512) | e(512)]
// ------------------------------------------------------------------
__global__ __launch_bounds__(256) void lstm_pointwise_kernel(
    const float* __restrict__ G,
    float* __restrict__ c,
    __bf16* __restrict__ hbf,
    __bf16* __restrict__ out2, int out2_stride)
{
    const int idx = blockIdx.x * blockDim.x + threadIdx.x; // over BSZ*HID
    const int b = idx >> 9;
    const int j = idx & (HID - 1);
    const float* g = G + (size_t)b * NGATE;
    const float xi = g[j];
    const float xf = g[HID + j];
    const float xg = g[2 * HID + j];
    const float xo = g[3 * HID + j];
    const float xe = g[4 * HID + j];
    const float i_ = 1.0f / (1.0f + __expf(-xi));
    const float f_ = 1.0f / (1.0f + __expf(-xf));
    const float g_ = tanhf(xg);
    const float o_ = 1.0f / (1.0f + __expf(-xo));
    const float e_ = __expf(tanhf(xe));
    const float cn = f_ * c[idx] + i_ * g_;
    c[idx] = cn;
    const float h = o_ * tanhf(cn) * e_;
    hbf[idx] = (__bf16)h;
    if (out2) out2[(size_t)b * out2_stride + j] = (__bf16)h;
}

// embedding gather -> time-major bf16 [S, B, E]
__global__ __launch_bounds__(256) void embed_kernel(
    const int* __restrict__ x, const float* __restrict__ emb, __bf16* __restrict__ out)
{
    const int idx = blockIdx.x * 256 + threadIdx.x;   // t<<16 | b<<8 | e
    const int e = idx & (EMB - 1);
    const int b = (idx >> 8) & (BSZ - 1);
    const int t = idx >> 16;
    const int tok = x[b * SEQ + t];
    out[idx] = (__bf16)emb[tok * EMB + e];
}

// pack [Wx | Wh ; We] -> bf16 [2560, K] row-major
__global__ __launch_bounds__(256) void pack_weights_kernel(
    const float* __restrict__ Wx, const float* __restrict__ Wh,
    const float* __restrict__ We, int in_d, int K, __bf16* __restrict__ out)
{
    const int idx = blockIdx.x * 256 + threadIdx.x;
    if (idx >= NGATE * K) return;
    const int row = idx / K, k = idx - row * K;
    float v;
    if (row < 4 * HID) v = (k < in_d) ? Wx[row * in_d + k] : Wh[row * HID + (k - in_d)];
    else               v = We[(size_t)(row - 4 * HID) * K + k];
    out[idx] = (__bf16)v;
}

__global__ __launch_bounds__(256) void pack_bias_kernel(
    const float* __restrict__ bx, const float* __restrict__ bh,
    const float* __restrict__ be, float* __restrict__ out)
{
    const int j = blockIdx.x * 256 + threadIdx.x;   // 2560 total
    if (j >= NGATE) return;
    out[j] = (j < 4 * HID) ? (bx[j] + bh[j]) : be[j - 4 * HID];
}

__global__ __launch_bounds__(256) void f2bf_kernel(
    const float* __restrict__ in, __bf16* __restrict__ out, int n)
{
    const int i = blockIdx.x * 256 + threadIdx.x;
    if (i < n) out[i] = (__bf16)in[i];
}

// ------------------------------------------------------------------
extern "C" void kernel_launch(void* const* d_in, const int* in_sizes, int n_in,
                              void* d_out, int out_size, void* d_ws, size_t ws_size,
                              hipStream_t stream) {
    (void)in_sizes; (void)n_in; (void)out_size; (void)ws_size;
    const int*   x    = (const int*)  d_in[0];
    const float* emb  = (const float*)d_in[1];
    const float* Wx0  = (const float*)d_in[2];
    const float* bx0  = (const float*)d_in[3];
    const float* Wh0  = (const float*)d_in[4];
    const float* bh0  = (const float*)d_in[5];
    const float* We0  = (const float*)d_in[6];
    const float* be0  = (const float*)d_in[7];
    const float* Wx1  = (const float*)d_in[8];
    const float* bx1  = (const float*)d_in[9];
    const float* Wh1  = (const float*)d_in[10];
    const float* bh1  = (const float*)d_in[11];
    const float* We1  = (const float*)d_in[12];
    const float* be1  = (const float*)d_in[13];
    const float* Wout = (const float*)d_in[14];
    const float* bout = (const float*)d_in[15];

    // ---- carve workspace (256B aligned chunks) ----
    char* ws = (char*)d_ws;
    auto carve = [&](size_t bytes) -> char* {
        char* p = ws; ws += (bytes + 255) & ~(size_t)255; return p;
    };
    const int K0L0 = EMB + HID;   // 768
    const int K0L1 = HID + HID;   // 1024
    __bf16* W0    = (__bf16*)carve((size_t)NGATE * K0L0 * 2);
    __bf16* W1    = (__bf16*)carve((size_t)NGATE * K0L1 * 2);
    __bf16* Wo    = (__bf16*)carve((size_t)VOC * HID * 2);
    float*  bias0 = (float*) carve((size_t)NGATE * 4);
    float*  bias1 = (float*) carve((size_t)NGATE * 4);
    __bf16* embbf = (__bf16*)carve((size_t)SEQ * BSZ * EMB * 2);
    __bf16* h0    = (__bf16*)carve((size_t)BSZ * HID * 2);
    __bf16* h1    = (__bf16*)carve((size_t)BSZ * HID * 2);
    float*  c0    = (float*) carve((size_t)BSZ * HID * 4);
    float*  c1    = (float*) carve((size_t)BSZ * HID * 4);
    float*  Gbuf  = (float*) carve((size_t)BSZ * NGATE * 4);
    __bf16* outs  = (__bf16*)carve((size_t)BSZ * SEQ * HID * 2);

    // ---- init recurrent state ----
    hipMemsetAsync(h0, 0, (size_t)BSZ * HID * 2, stream);
    hipMemsetAsync(h1, 0, (size_t)BSZ * HID * 2, stream);
    hipMemsetAsync(c0, 0, (size_t)BSZ * HID * 4, stream);
    hipMemsetAsync(c1, 0, (size_t)BSZ * HID * 4, stream);

    // ---- one-time packing ----
    pack_weights_kernel<<<(NGATE * K0L0 + 255) / 256, 256, 0, stream>>>(Wx0, Wh0, We0, EMB, K0L0, W0);
    pack_weights_kernel<<<(NGATE * K0L1 + 255) / 256, 256, 0, stream>>>(Wx1, Wh1, We1, HID, K0L1, W1);
    pack_bias_kernel<<<(NGATE + 255) / 256, 256, 0, stream>>>(bx0, bh0, be0, bias0);
    pack_bias_kernel<<<(NGATE + 255) / 256, 256, 0, stream>>>(bx1, bh1, be1, bias1);
    f2bf_kernel<<<(VOC * HID + 255) / 256, 256, 0, stream>>>(Wout, Wo, VOC * HID);
    embed_kernel<<<(SEQ * BSZ * EMB) / 256, 256, 0, stream>>>(x, emb, embbf);

    // ---- recurrence: per step, one fused GEMM + pointwise per layer ----
    dim3 gs(NGATE / BN, BSZ / BM);          // (20, 2)
    for (int t = 0; t < SEQ; ++t) {
        const __bf16* xt = embbf + (size_t)t * BSZ * EMB;
        // layer 0: A = [x_t(E) | h0(H)]
        gemm_bf16_kernel<<<gs, 256, 0, stream>>>(xt, EMB, EMB, h0, HID,
                                                 W0, K0L0, bias0, Gbuf, NGATE, K0L0);
        lstm_pointwise_kernel<<<(BSZ * HID) / 256, 256, 0, stream>>>(Gbuf, c0, h0,
                                                                     (__bf16*)nullptr, 0);
        // layer 1: A = [h0(H) | h1(H)]
        gemm_bf16_kernel<<<gs, 256, 0, stream>>>(h0, HID, HID, h1, HID,
                                                 W1, K0L1, bias1, Gbuf, NGATE, K0L1);
        lstm_pointwise_kernel<<<(BSZ * HID) / 256, 256, 0, stream>>>(Gbuf, c1, h1,
                                                                     outs + (size_t)t * HID,
                                                                     SEQ * HID);
    }

    // ---- vocab projection: logits[B*S, V] = outs @ Wout^T + bout ----
    dim3 go(VOC / BN, (BSZ * SEQ) / BM);    // (8, 512)
    gemm_bf16_kernel<<<go, 256, 0, stream>>>(outs, HID, HID, outs, HID,
                                             Wo, HID, bout, (float*)d_out, VOC, HID);
}
